// TacoSinger_51780125721062
// MI455X (gfx1250) — matrix operand order repaired and checked by reading, not compiled, not run on hardware
//
#include <hip/hip_runtime.h>
#include <hip/hip_bf16.h>
#include <stdint.h>

#define B_      16
#define ENC_DD  512
#define ENC_TT  256
#define FEATD   513
#define DECT    512
#define PRE2_   256
#define NWG     32

typedef __attribute__((ext_vector_type(16))) __bf16 v16bf;
typedef __attribute__((ext_vector_type(8)))  float  v8f;

union BfFrag { v16bf v; uint4 q[2]; unsigned short u[16]; };

__device__ __forceinline__ unsigned short f2bf(float x) {
  union { float f; unsigned u; } w; w.f = x;
  unsigned r = w.u + 0x7fffu + ((w.u >> 16) & 1u);
  return (unsigned short)(r >> 16);
}

// A fragment: 16x32 bf16, rows m = lane&15, K split per ISA layout
__device__ __forceinline__ v16bf load_a_frag(const unsigned short* base, int stride, int k0, int lane) {
  int m = lane & 15, kh = lane >> 4;
  const unsigned short* p = base + (size_t)m * stride + k0 + kh * 8;
  BfFrag f;
  f.q[0] = *(const uint4*)(p);        // K = k0 + kh*8 + 0..7
  f.q[1] = *(const uint4*)(p + 16);   // K = k0 + 16 + kh*8 + 0..7
  return f.v;
}

// B fragment: 32x16 bf16 from row-major weight W[N][K] (gates = x @ W^T)
__device__ __forceinline__ v16bf load_b_frag(const unsigned short* w, int ldk, int n0, int k0, int lane) {
  const unsigned short* p = w + (size_t)(n0 + (lane & 15)) * ldk + k0 + (lane >> 4) * 8;
  BfFrag f;
  f.q[0] = *(const uint4*)(p);
  f.q[1] = *(const uint4*)(p + 16);
  return f.v;
}

__device__ __forceinline__ v8f wmma_bf(v16bf a, v16bf b, v8f c) {
  return __builtin_amdgcn_wmma_f32_16x16x32_bf16(false, a, false, b, (short)0, c, false, false);
}

__device__ __forceinline__ float sigm(float x) { return 1.f / (1.f + __expf(-x)); }

__device__ void grid_barrier(unsigned* cnt, unsigned* gen) {
  __threadfence();
  __syncthreads();
  if (threadIdx.x == 0) {
    unsigned my = atomicAdd(gen, 0u);
    if (atomicAdd(cnt, 1u) == NWG - 1) {
      atomicExch(cnt, 0u);
      atomicAdd(gen, 1u);
    } else {
      while (atomicAdd(gen, 0u) == my) { __builtin_amdgcn_s_sleep(2); }
    }
  }
  __syncthreads();
  __threadfence();
}

// ---------------- preprocessing kernels ----------------

// f32 -> bf16 with optional zero-padding of rows/cols
__global__ void cvt_bf16(const float* __restrict__ src, unsigned short* __restrict__ dst,
                         int rows, int cols, int drows, int dcols) {
  int idx = blockIdx.x * 256 + threadIdx.x;
  int total = drows * dcols;
  if (idx >= total) return;
  int r = idx / dcols, c = idx - r * dcols;
  float v = (r < rows && c < cols) ? src[(size_t)r * cols + c] : 0.f;
  dst[idx] = f2bf(v);
}

// dec_in rows ordered [t][b], K padded 513 -> 544
__global__ void build_decin(const float* __restrict__ features, const float* __restrict__ go,
                            unsigned short* __restrict__ dst) {
  int idx = blockIdx.x * 256 + threadIdx.x;
  const int total = DECT * B_ * 544;
  if (idx >= total) return;
  int k = idx % 544;
  int row = idx / 544;
  int b = row & 15, t = row >> 4;
  float v = 0.f;
  if (k < FEATD) v = (t == 0) ? go[k] : features[(size_t)b * FEATD * DECT + (size_t)k * DECT + (t - 1)];
  dst[idx] = f2bf(v);
}

// generic 16-row-tile WMMA gemm: out = relu(A[MxK] * W[NxK]^T) in bf16
__global__ void prenet_gemm(const unsigned short* __restrict__ A, const unsigned short* __restrict__ W,
                            unsigned short* __restrict__ out, int M, int N, int K) {
  int wid = (blockIdx.x * 256 + threadIdx.x) >> 5;
  int lane = threadIdx.x & 31;
  int Nt = N >> 4;
  int total = (M >> 4) * Nt;
  if (wid >= total) return;
  int mt = wid / Nt, nt = wid - mt * Nt;
  const unsigned short* Abase = A + (size_t)mt * 16 * K;
  v8f c = {};
  for (int k0 = 0; k0 < K; k0 += 32) {
    v16bf a = load_a_frag(Abase, K, k0, lane);
    v16bf b = load_b_frag(W, K, nt * 16, k0, lane);
    c = wmma_bf(a, b, c);
  }
  int n = nt * 16 + (lane & 15);
  int mh = (lane >> 4) * 8;
#pragma unroll
  for (int j = 0; j < 8; ++j) {
    float v = c[j]; if (v < 0.f) v = 0.f;
    out[(size_t)(mt * 16 + mh + j) * N + n] = f2bf(v);
  }
}

// proc_mem[b][t][a] = sum_d enc[b][d][t] * Wm[a][d]  (fp32, one-off)
__global__ void proc_mem_k(const float* __restrict__ enc, const float* __restrict__ wm,
                           float* __restrict__ pm) {
  int idx = blockIdx.x * 256 + threadIdx.x;
  if (idx >= B_ * ENC_TT * 128) return;
  int a = idx & 127;
  int bt = idx >> 7;
  int t = bt & 255, b = bt >> 8;
  float s = 0.f;
  for (int d = 0; d < ENC_DD; ++d)
    s += enc[((size_t)b * ENC_DD + d) * ENC_TT + t] * wm[(size_t)a * ENC_DD + d];
  pm[idx] = s;
}

__global__ void init_state(float* c1, float* c2, float* h1,
                           unsigned short* h1bf, unsigned short* h2bf, unsigned short* ctxbf,
                           float* align, float* cum, unsigned* bar) {
  int i = blockIdx.x * 256 + threadIdx.x;
  if (i < B_ * 1024) { c1[i] = 0.f; c2[i] = 0.f; h1[i] = 0.f; h1bf[i] = 0; h2bf[i] = 0; }
  if (i < B_ * 512) ctxbf[i] = 0;
  if (i < B_ * ENC_TT) {
    float v = ((i & 255) == 0) ? 1.f : 0.f;   // align0 one-hot; cum starts as align0
    align[i] = v; cum[i] = v;
  }
  if (i < 2) bar[i] = 0u;
}

// ---------------- persistent decoder ----------------

struct DecP {
  const unsigned short *w1ih, *w1hh, *w2ih, *w2hh, *wproj;   // bf16 weights (row-major [N][K])
  const float *pre_b, *post_b, *proj_b;
  const unsigned short *prenet;     // [512][16][256] bf16
  const float *pm;                  // [16][256][128]
  const float *enc;                 // [16][512][256]
  const int   *enc_len;
  const float *wq, *wconv, *wld, *attv;
  float *c1, *c2, *h1;
  unsigned short *h1bf, *h2bf, *ctxbf;
  float *align, *cum, *g2p;
  unsigned *bar;
  float *out_feat, *out_align;
};

__global__ __launch_bounds__(256) void decoder(DecP p) {
  const int tid  = threadIdx.x;
  const int lane = tid & 31;
  const int wv   = tid >> 5;
  const int wg   = blockIdx.x;
  const int gwave = wg * 8 + wv;

  __shared__ float s_g[4][16][32];      // gate tiles for this WG's 32 columns
  __shared__ float s_pq[128];
  __shared__ float s_lfc[32][256];
  __shared__ float s_al[256];
  __shared__ float s_cm[256];
  __shared__ float s_e[256];
  __shared__ float s_red[256];

#pragma unroll 1
  for (int t = 0; t < DECT; ++t) {
    // ======== Phase A: proj(t-1) + LSTM1 gates + pointwise ========
    if (t > 0 && gwave < 33) {
      int n0 = gwave * 16;
      v8f c = {};
      for (int kt = 0; kt < 48; ++kt) {
        int k0 = kt * 32;
        v16bf a = (k0 < 1024) ? load_a_frag(p.h2bf, 1024, k0, lane)
                              : load_a_frag(p.ctxbf, 512, k0 - 1024, lane);
        v16bf b = load_b_frag(p.wproj, 1536, n0, k0, lane);
        c = wmma_bf(a, b, c);
      }
      int n = n0 + (lane & 15);
      if (n < FEATD) {
        int mh = (lane >> 4) * 8;
        float bias = p.proj_b[n];
#pragma unroll
        for (int j = 0; j < 8; ++j)
          p.out_feat[(size_t)(mh + j) * FEATD * DECT + (size_t)n * DECT + (t - 1)] = c[j] + bias;
      }
    }
    {
      int gi = wv >> 1;
      int coff = (wv & 1) * 16;
      int n0 = gi * 1024 + wg * 32 + coff;
      const unsigned short* preA = p.prenet + (size_t)t * B_ * PRE2_;
      v8f c = {};
      for (int kt = 0; kt < 8; ++kt) {              // prenet part of x (K 0..255)
        v16bf a = load_a_frag(preA, PRE2_, kt * 32, lane);
        v16bf b = load_b_frag(p.w1ih, 768, n0, kt * 32, lane);
        c = wmma_bf(a, b, c);
      }
      for (int kt = 0; kt < 16; ++kt) {             // ctx part (K 256..767)
        v16bf a = load_a_frag(p.ctxbf, 512, kt * 32, lane);
        v16bf b = load_b_frag(p.w1ih, 768, n0, 256 + kt * 32, lane);
        c = wmma_bf(a, b, c);
      }
      for (int kt = 0; kt < 32; ++kt) {             // h1 @ W_hh
        v16bf a = load_a_frag(p.h1bf, 1024, kt * 32, lane);
        v16bf b = load_b_frag(p.w1hh, 1024, n0, kt * 32, lane);
        c = wmma_bf(a, b, c);
      }
      int n = lane & 15, mh = (lane >> 4) * 8;
#pragma unroll
      for (int j = 0; j < 8; ++j) s_g[gi][mh + j][coff + n] = c[j];
    }
    __syncthreads();
    for (int idx = tid; idx < 512; idx += 256) {
      int m = idx >> 5, cc = idx & 31, col = wg * 32 + cc;
      float gI = s_g[0][m][cc] + p.pre_b[col];
      float gF = s_g[1][m][cc] + p.pre_b[1024 + col];
      float gG = s_g[2][m][cc] + p.pre_b[2048 + col];
      float gO = s_g[3][m][cc] + p.pre_b[3072 + col];
      float cn = sigm(gF) * p.c1[m * 1024 + col] + sigm(gI) * tanhf(gG);
      p.c1[m * 1024 + col] = cn;
      float h = sigm(gO) * tanhf(cn);
      p.h1[m * 1024 + col] = h;
      p.h1bf[m * 1024 + col] = f2bf(h);
    }
    grid_barrier(p.bar, p.bar + 1);

    // ======== Phase B: attention (WG 0..15) || LSTM2 partial gates (WG 16..31) ========
    if (wg < B_) {
      int b = wg;
      int len = p.enc_len[b];
      s_al[tid] = p.align[b * ENC_TT + tid];
      s_cm[tid] = p.cum[b * ENC_TT + tid];
      if (tid < 128) {
        float s = 0.f;
        for (int k = 0; k < 1024; ++k) s += p.h1[b * 1024 + k] * p.wq[(size_t)tid * 1024 + k];
        s_pq[tid] = s;
      }
      __syncthreads();
      // location conv (cross-correlation, pad 15)
      for (int c = 0; c < 32; ++c) {
        float s = 0.f;
        for (int j = 0; j < 31; ++j) {
          int src = tid + j - 15;
          if (src >= 0 && src < ENC_TT)
            s += p.wconv[c * 62 + j] * s_al[src] + p.wconv[c * 62 + 31 + j] * s_cm[src];
        }
        s_lfc[c][tid] = s;
      }
      __syncthreads();
      // energies
      {
        float e = 0.f;
        const float* pmrow = p.pm + ((size_t)b * ENC_TT + tid) * 128;
        for (int a = 0; a < 128; ++a) {
          float lf = 0.f;
          for (int c = 0; c < 32; ++c) lf += s_lfc[c][tid] * p.wld[a * 32 + c];
          e += p.attv[a] * tanhf(s_pq[a] + pmrow[a] + lf);
        }
        if (tid >= len) e = -1e9f;
        s_e[tid] = e;
        s_red[tid] = e;
      }
      __syncthreads();
      for (int st = 128; st > 0; st >>= 1) {
        if (tid < st) s_red[tid] = fmaxf(s_red[tid], s_red[tid + st]);
        __syncthreads();
      }
      float mx = s_red[0];
      __syncthreads();
      float pex = __expf(s_e[tid] - mx);
      s_red[tid] = pex;
      __syncthreads();
      for (int st = 128; st > 0; st >>= 1) {
        if (tid < st) s_red[tid] += s_red[tid + st];
        __syncthreads();
      }
      float aw = pex / s_red[0];
      s_e[tid] = aw;
      p.align[b * ENC_TT + tid] = aw;
      p.cum[b * ENC_TT + tid] = s_cm[tid] + aw;
      p.out_align[(size_t)b * ENC_TT * DECT + (size_t)tid * DECT + t] = aw;
      __syncthreads();
      // ctx = a @ memories
      for (int d = tid; d < ENC_DD; d += 256) {
        float s = 0.f;
        const float* erow = p.enc + ((size_t)b * ENC_DD + d) * ENC_TT;
        for (int t2 = 0; t2 < ENC_TT; ++t2) s += s_e[t2] * erow[t2];
        p.ctxbf[b * ENC_DD + d] = f2bf(s);
      }
    } else {
      int q = wg - B_;
      for (int tt2 = 0; tt2 < 2; ++tt2) {
        int n0 = q * 256 + (wv * 2 + tt2) * 16;
        v8f c = {};
        for (int kt = 0; kt < 32; ++kt) {           // h2(t-1) @ post_w_hh
          v16bf a = load_a_frag(p.h2bf, 1024, kt * 32, lane);
          v16bf b = load_b_frag(p.w2hh, 1024, n0, kt * 32, lane);
          c = wmma_bf(a, b, c);
        }
        for (int kt = 0; kt < 32; ++kt) {           // h1(t) @ post_w_ih[:, :1024]
          v16bf a = load_a_frag(p.h1bf, 1024, kt * 32, lane);
          v16bf b = load_b_frag(p.w2ih, 1536, n0, kt * 32, lane);
          c = wmma_bf(a, b, c);
        }
        int n = n0 + (lane & 15), mh = (lane >> 4) * 8;
#pragma unroll
        for (int j = 0; j < 8; ++j) p.g2p[(size_t)(mh + j) * 4096 + n] = c[j];
      }
    }
    grid_barrier(p.bar, p.bar + 1);

    // ======== Phase C: finish LSTM2 (ctx slice) + pointwise ========
    {
      int gi = wv >> 1;
      int coff = (wv & 1) * 16;
      int n0 = gi * 1024 + wg * 32 + coff;
      v8f c;
      {
        int n = n0 + (lane & 15), mh = (lane >> 4) * 8;
#pragma unroll
        for (int j = 0; j < 8; ++j) c[j] = p.g2p[(size_t)(mh + j) * 4096 + n];
      }
      for (int kt = 0; kt < 16; ++kt) {             // ctx(t) @ post_w_ih[:, 1024:]
        v16bf a = load_a_frag(p.ctxbf, 512, kt * 32, lane);
        v16bf b = load_b_frag(p.w2ih, 1536, n0, 1024 + kt * 32, lane);
        c = wmma_bf(a, b, c);
      }
      int n = lane & 15, mh = (lane >> 4) * 8;
#pragma unroll
      for (int j = 0; j < 8; ++j) s_g[gi][mh + j][coff + n] = c[j];
    }
    __syncthreads();
    for (int idx = tid; idx < 512; idx += 256) {
      int m = idx >> 5, cc = idx & 31, col = wg * 32 + cc;
      float gI = s_g[0][m][cc] + p.post_b[col];
      float gF = s_g[1][m][cc] + p.post_b[1024 + col];
      float gG = s_g[2][m][cc] + p.post_b[2048 + col];
      float gO = s_g[3][m][cc] + p.post_b[3072 + col];
      float cn = sigm(gF) * p.c2[m * 1024 + col] + sigm(gI) * tanhf(gG);
      p.c2[m * 1024 + col] = cn;
      float h = sigm(gO) * tanhf(cn);
      p.h2bf[m * 1024 + col] = f2bf(h);
    }
    grid_barrier(p.bar, p.bar + 1);
  }

  // final projection for t = DECT-1
  if (gwave < 33) {
    int n0 = gwave * 16;
    v8f c = {};
    for (int kt = 0; kt < 48; ++kt) {
      int k0 = kt * 32;
      v16bf a = (k0 < 1024) ? load_a_frag(p.h2bf, 1024, k0, lane)
                            : load_a_frag(p.ctxbf, 512, k0 - 1024, lane);
      v16bf b = load_b_frag(p.wproj, 1536, n0, k0, lane);
      c = wmma_bf(a, b, c);
    }
    int n = n0 + (lane & 15);
    if (n < FEATD) {
      int mh = (lane >> 4) * 8;
      float bias = p.proj_b[n];
#pragma unroll
      for (int j = 0; j < 8; ++j)
        p.out_feat[(size_t)(mh + j) * FEATD * DECT + (size_t)n * DECT + (DECT - 1)] = c[j] + bias;
    }
  }
}

// ---------------- host launch ----------------

extern "C" void kernel_launch(void* const* d_in, const int* in_sizes, int n_in,
                              void* d_out, int out_size, void* d_ws, size_t ws_size,
                              hipStream_t stream) {
  (void)in_sizes; (void)n_in; (void)out_size; (void)ws_size;
  const float* enc     = (const float*)d_in[0];
  const int*   enclen  = (const int*)d_in[1];
  const float* feats   = (const float*)d_in[2];
  const float* go      = (const float*)d_in[3];
  const float* w0f     = (const float*)d_in[4];
  const float* w1f     = (const float*)d_in[5];
  const float* w1ihf   = (const float*)d_in[6];
  const float* w1hhf   = (const float*)d_in[7];
  const float* pre_b   = (const float*)d_in[8];
  const float* w2ihf   = (const float*)d_in[9];
  const float* w2hhf   = (const float*)d_in[10];
  const float* post_b  = (const float*)d_in[11];
  const float* wq      = (const float*)d_in[12];
  const float* wm      = (const float*)d_in[13];
  const float* wconv   = (const float*)d_in[14];
  const float* wld     = (const float*)d_in[15];
  const float* attv    = (const float*)d_in[16];
  const float* wprojf  = (const float*)d_in[17];
  const float* proj_b  = (const float*)d_in[18];

  size_t ofs = 0;
  auto alloc = [&](size_t bytes) {
    void* ptr = (char*)d_ws + ofs;
    ofs += (bytes + 255) & ~(size_t)255;
    return ptr;
  };
  unsigned short* w1ih_bf  = (unsigned short*)alloc((size_t)4096 * 768 * 2);
  unsigned short* w1hh_bf  = (unsigned short*)alloc((size_t)4096 * 1024 * 2);
  unsigned short* w2ih_bf  = (unsigned short*)alloc((size_t)4096 * 1536 * 2);
  unsigned short* w2hh_bf  = (unsigned short*)alloc((size_t)4096 * 1024 * 2);
  unsigned short* wproj_bf = (unsigned short*)alloc((size_t)528 * 1536 * 2);
  unsigned short* w0_bf    = (unsigned short*)alloc((size_t)256 * 544 * 2);
  unsigned short* w1p_bf   = (unsigned short*)alloc((size_t)256 * 256 * 2);
  unsigned short* decin_bf = (unsigned short*)alloc((size_t)8192 * 544 * 2);
  unsigned short* pre1_bf  = (unsigned short*)alloc((size_t)8192 * 256 * 2);
  unsigned short* pren_bf  = (unsigned short*)alloc((size_t)8192 * 256 * 2);
  float* pm    = (float*)alloc((size_t)16 * 256 * 128 * 4);
  float* c1    = (float*)alloc((size_t)16 * 1024 * 4);
  float* c2    = (float*)alloc((size_t)16 * 1024 * 4);
  float* h1    = (float*)alloc((size_t)16 * 1024 * 4);
  unsigned short* h1bf  = (unsigned short*)alloc((size_t)16 * 1024 * 2);
  unsigned short* h2bf  = (unsigned short*)alloc((size_t)16 * 1024 * 2);
  unsigned short* ctxbf = (unsigned short*)alloc((size_t)16 * 512 * 2);
  float* align = (float*)alloc((size_t)16 * 256 * 4);
  float* cum   = (float*)alloc((size_t)16 * 256 * 4);
  float* g2p   = (float*)alloc((size_t)16 * 4096 * 4);
  unsigned* bar = (unsigned*)alloc(2 * sizeof(unsigned));

  auto blocks = [](size_t n) { return (unsigned)((n + 255) / 256); };

  // weight conversions (fp32 -> bf16, padded where needed)
  cvt_bf16<<<blocks((size_t)4096*768),  256, 0, stream>>>(w1ihf, w1ih_bf, 4096, 768, 4096, 768);
  cvt_bf16<<<blocks((size_t)4096*1024), 256, 0, stream>>>(w1hhf, w1hh_bf, 4096, 1024, 4096, 1024);
  cvt_bf16<<<blocks((size_t)4096*1536), 256, 0, stream>>>(w2ihf, w2ih_bf, 4096, 1536, 4096, 1536);
  cvt_bf16<<<blocks((size_t)4096*1024), 256, 0, stream>>>(w2hhf, w2hh_bf, 4096, 1024, 4096, 1024);
  cvt_bf16<<<blocks((size_t)528*1536),  256, 0, stream>>>(wprojf, wproj_bf, 513, 1536, 528, 1536);
  cvt_bf16<<<blocks((size_t)256*544),   256, 0, stream>>>(w0f, w0_bf, 256, 513, 256, 544);
  cvt_bf16<<<blocks((size_t)256*256),   256, 0, stream>>>(w1f, w1p_bf, 256, 256, 256, 256);

  build_decin<<<blocks((size_t)8192*544), 256, 0, stream>>>(feats, go, decin_bf);
  init_state<<<64, 256, 0, stream>>>(c1, c2, h1, h1bf, h2bf, ctxbf, align, cum, bar);

  // prenet: relu(relu(decin @ W0^T) @ W1^T), all 8192 rows in parallel via WMMA
  prenet_gemm<<<1024, 256, 0, stream>>>(decin_bf, w0_bf, pre1_bf, 8192, 256, 544);
  prenet_gemm<<<1024, 256, 0, stream>>>(pre1_bf, w1p_bf, pren_bf, 8192, 256, 256);

  proc_mem_k<<<blocks((size_t)16*256*128), 256, 0, stream>>>(enc, wm, pm);

  DecP p;
  p.w1ih = w1ih_bf; p.w1hh = w1hh_bf; p.w2ih = w2ih_bf; p.w2hh = w2hh_bf; p.wproj = wproj_bf;
  p.pre_b = pre_b; p.post_b = post_b; p.proj_b = proj_b;
  p.prenet = pren_bf; p.pm = pm; p.enc = enc; p.enc_len = enclen;
  p.wq = wq; p.wconv = wconv; p.wld = wld; p.attv = attv;
  p.c1 = c1; p.c2 = c2; p.h1 = h1;
  p.h1bf = h1bf; p.h2bf = h2bf; p.ctxbf = ctxbf;
  p.align = align; p.cum = cum; p.g2p = g2p; p.bar = bar;
  p.out_feat = (float*)d_out;
  p.out_align = (float*)d_out + (size_t)B_ * FEATD * DECT;

  decoder<<<NWG, 256, 0, stream>>>(p);
}
